// GraphMP_4690104287811
// MI455X (gfx1250) — matrix operand order repaired and checked by reading, not compiled, not run on hardware
//
#include <hip/hip_runtime.h>

#define N_ 10000
#define E_ 100000
#define G_ 64
#define NTILES 6250          // E_/16 exactly
#define EPS_ 1e-5f

typedef __attribute__((ext_vector_type(16))) _Float16 v16h;
typedef __attribute__((ext_vector_type(8)))  float    v8f;

union HU { unsigned u; _Float16 h[2]; };

__device__ __forceinline__ unsigned pk_f16(float a, float b) {
    HU x; x.h[0] = (_Float16)a; x.h[1] = (_Float16)b; return x.u;
}

// -------- zero scratch --------
__global__ void k_zero(float* __restrict__ p, int n) {
    int i = blockIdx.x * blockDim.x + threadIdx.x;
    if (i < n) p[i] = 0.f;
}

// -------- in-degree + pool counts --------
__global__ void k_counts(const int* __restrict__ dst, const int* __restrict__ batch,
                         float* __restrict__ cnt, float* __restrict__ pcnt) {
    int i = blockIdx.x * blockDim.x + threadIdx.x;
    if (i < E_) atomicAdd(&cnt[dst[i]], 1.0f);
    if (i < N_) atomicAdd(&pcnt[batch[i]], 1.0f);
}

// -------- pack w2/b2 into WMMA B-fragment order (f16) --------
// layout: [layer][frag=c*2+nblk (c 0..32)][lane 0..31][dword 0..7]
// B 32x16 f16 lane layout: lane<16: N=lane, Vd = K(2d,2d+1) for d<4, K(8+2d,9+2d) for d>=4; lane>=16: N=lane-16, Vd = K(8+2d,9+2d) for d<4, K(16+2d,17+2d) for d>=4
__global__ void k_pack(const float* __restrict__ w2a, const float* __restrict__ b2a,
                       const float* __restrict__ w2b, const float* __restrict__ b2b,
                       unsigned* __restrict__ pB) {
    int gid = blockIdx.x * blockDim.x + threadIdx.x;
    if (gid >= 2 * 16896) return;
    int layer = gid / 16896;
    int r     = gid - layer * 16896;
    int frag  = r >> 8;          // /256
    int lid   = (r >> 3) & 31;
    int d     = r & 7;
    const float* w2 = layer ? w2b : w2a;
    const float* b2 = layer ? b2b : b2a;
    int c     = frag >> 1;
    int nblk  = frag & 1;
    int n     = lid & 15;
    int kbase = (lid >= 16) ? 8 : 0;
    int k0    = kbase + 2 * d + (d >= 4 ? 8 : 0);
    int o     = nblk * 16 + n;
    float v0, v1;
    if (c < 32) { v0 = w2[k0 * 1024 + c * 32 + o]; v1 = w2[(k0 + 1) * 1024 + c * 32 + o]; }
    else        { v0 = b2[k0 * 32 + o];            v1 = b2[(k0 + 1) * 32 + o]; }
    pB[gid] = pk_f16(v0, v1);
}

// -------- edge MLPs (both layers): h = relu(ea @ w1 + b1), stored as f16 pairs --------
__global__ void k_edge_h(const float* __restrict__ ea,
                         const float* __restrict__ w1a, const float* __restrict__ b1a,
                         const float* __restrict__ w1b, const float* __restrict__ b1b,
                         unsigned* __restrict__ h1, unsigned* __restrict__ h2) {
    int e = blockIdx.x * blockDim.x + threadIdx.x;
    if (e >= E_) return;
    float a0 = ea[e * 4 + 0], a1 = ea[e * 4 + 1], a2 = ea[e * 4 + 2], a3 = ea[e * 4 + 3];
    const float* W[2] = { w1a, w1b };
    const float* B[2] = { b1a, b1b };
    unsigned*    H[2] = { h1, h2 };
    #pragma unroll
    for (int L = 0; L < 2; ++L) {
        const float* w = W[L]; const float* b = B[L]; unsigned* hh = H[L];
        #pragma unroll
        for (int j = 0; j < 32; j += 2) {
            float u0 = fmaxf(fmaf(a3, w[96 + j],     fmaf(a2, w[64 + j],     fmaf(a1, w[32 + j],     fmaf(a0, w[j],     b[j])))),     0.f);
            float u1 = fmaxf(fmaf(a3, w[96 + j + 1], fmaf(a2, w[64 + j + 1], fmaf(a1, w[32 + j + 1], fmaf(a0, w[j + 1], b[j + 1])))), 0.f);
            hh[(size_t)e * 16 + (j >> 1)] = pk_f16(u0, u1);
        }
    }
}

// -------- NNConv GEMM+scatter: msg = P @ W2' (+ x@b2) via WMMA, atomic scatter to agg --------
__global__ void __launch_bounds__(256)
k_nnconv(const float* __restrict__ x, const unsigned* __restrict__ hpk,
         const unsigned* __restrict__ pB, const int* __restrict__ ei,
         float* __restrict__ agg) {
    extern __shared__ unsigned ldsB[];               // 66 frags * 256 dwords = 67584 B
    {   // staged copy, vectorized b128: 16896 dwords = 4224 uint4
        uint4*       dstv = (uint4*)ldsB;
        const uint4* srcv = (const uint4*)pB;
        for (int i = threadIdx.x; i < 4224; i += blockDim.x) dstv[i] = srcv[i];
    }
    __syncthreads();

    const int lane = threadIdx.x & 31;
    const int hi16 = lane >> 4;
    const int m    = lane & 15;
    const int wv   = blockIdx.x * (blockDim.x >> 5) + (threadIdx.x >> 5);
    const int nwv  = gridDim.x * (blockDim.x >> 5);
    const int* srcI = ei;
    const int* dstI = ei + E_;
    const int kb0 = hi16 ? 8 : 0, kb1 = hi16 ? 24 : 16;

    for (int t = wv; t < NTILES; t += nwv) {
        // Memory clobber: LDS B fragments must NOT be treated as loop-invariant
        // (otherwise LICM hoists 66x16 VGPRs of fragments and spills to scratch).
        asm volatile("" ::: "memory");

        int e = t * 16 + m;
        int s = srcI[e];
        // full x row of this lane's edge-row (lanes L and L+16 share row m)
        float xs[32];
        const float4* xr = (const float4*)(x + (size_t)s * 32);
        #pragma unroll
        for (int q = 0; q < 8; ++q) {
            float4 f = xr[q];
            xs[4 * q] = f.x; xs[4 * q + 1] = f.y; xs[4 * q + 2] = f.z; xs[4 * q + 3] = f.w;
        }
        // lane's A-layout h subset: lane<16 -> K{0..7,16..23} (dw 0..3,8..11); lane>=16 -> K{8..15,24..31} (dw 4..7,12..15)
        float hf[16];
        const unsigned* hr = hpk + (size_t)e * 16 + (hi16 ? 4 : 0);
        #pragma unroll
        for (int q = 0; q < 4; ++q) { HU u; u.u = hr[q];     hf[2 * q]     = (float)u.h[0]; hf[2 * q + 1]     = (float)u.h[1]; }
        #pragma unroll
        for (int q = 0; q < 4; ++q) { HU u; u.u = hr[8 + q]; hf[8 + 2 * q] = (float)u.h[0]; hf[8 + 2 * q + 1] = (float)u.h[1]; }

        v8f c0 = {0.f,0.f,0.f,0.f,0.f,0.f,0.f,0.f};
        v8f c1 = {0.f,0.f,0.f,0.f,0.f,0.f,0.f,0.f};

        #pragma unroll
        for (int c = 0; c < 33; ++c) {             // K-chunk c == input channel i; c==32 is the b2 bias chunk
            v16h a;
            if (c < 32) {
                float sc = xs[c];
                #pragma unroll
                for (int j = 0; j < 16; ++j) a[j] = (_Float16)(sc * hf[j]);
            } else {
                #pragma unroll
                for (int j = 0; j < 8; ++j) a[j]     = (_Float16)xs[kb0 + j];
                #pragma unroll
                for (int j = 0; j < 8; ++j) a[8 + j] = (_Float16)xs[kb1 + j];
            }
            // Clobber before each fragment pair: keeps at most ~2 B fragments live,
            // so the unrolled K-loop can't front-load all ds reads and spill.
            asm volatile("" ::: "memory");
            // Fragment base = frag*1024 B + lane*32 B -> 32-byte aligned: allow ds_load_b128
            const v16h b0 = *(const v16h*)__builtin_assume_aligned(ldsB + (2 * c    ) * 256 + lane * 8, 32);
            const v16h b1 = *(const v16h*)__builtin_assume_aligned(ldsB + (2 * c + 1) * 256 + lane * 8, 32);
            c0 = __builtin_amdgcn_wmma_f32_16x16x32_f16(false, a, false, b0, (short)0, c0, false, false);
            c1 = __builtin_amdgcn_wmma_f32_16x16x32_f16(false, a, false, b1, (short)0, c1, false, false);
        }
        // C layout: VGPR v -> M = v + 8*hi16, N = lane%16 ; scatter-add to agg[dst]
        const int mb = hi16 ? 8 : 0;
        #pragma unroll
        for (int v = 0; v < 8; ++v) {
            int d = dstI[t * 16 + mb + v];
            atomicAdd(&agg[(size_t)d * 32 + m],      c0[v]);
            atomicAdd(&agg[(size_t)d * 32 + 16 + m], c1[v]);
        }
    }
}

// -------- node update: mean agg + x@root + bias, BN(eval), ReLU; optional graph-pool --------
__global__ void __launch_bounds__(256)
k_node(const float* __restrict__ agg, const float* __restrict__ cnt,
       const float* __restrict__ xin, const float* __restrict__ root,
       const float* __restrict__ bias,
       const float* __restrict__ bg, const float* __restrict__ bb,
       const float* __restrict__ bm, const float* __restrict__ bv,
       float* __restrict__ xout,
       const int* __restrict__ batch, float* __restrict__ psum, int doPool) {
    __shared__ float R[1024];
    for (int i = threadIdx.x; i < 1024; i += blockDim.x) R[i] = root[i];
    __syncthreads();
    int n = blockIdx.x * blockDim.x + threadIdx.x;
    if (n >= N_) return;
    float xr[32];
    const float4* xp = (const float4*)(xin + (size_t)n * 32);
    #pragma unroll
    for (int q = 0; q < 8; ++q) {
        float4 f = xp[q];
        xr[4 * q] = f.x; xr[4 * q + 1] = f.y; xr[4 * q + 2] = f.z; xr[4 * q + 3] = f.w;
    }
    float ic = 1.f / fmaxf(cnt[n], 1.f);
    int bidx = doPool ? batch[n] : 0;
    #pragma unroll 4
    for (int o = 0; o < 32; ++o) {
        float s = agg[(size_t)n * 32 + o] * ic + bias[o];
        #pragma unroll
        for (int i = 0; i < 32; ++i) s = fmaf(xr[i], R[i * 32 + o], s);
        s = (s - bm[o]) * rsqrtf(bv[o] + EPS_) * bg[o] + bb[o];
        s = fmaxf(s, 0.f);
        xout[(size_t)n * 32 + o] = s;
        if (doPool) atomicAdd(&psum[bidx * 32 + o], s);
    }
}

// -------- readout MLP over G graphs --------
__global__ void k_readout(const float* __restrict__ psum, const float* __restrict__ pcnt,
                          const float* __restrict__ r1w, const float* __restrict__ r1b,
                          const float* __restrict__ r2w, const float* __restrict__ r2b,
                          float* __restrict__ out) {
    int g = blockIdx.x * blockDim.x + threadIdx.x;
    if (g >= G_) return;
    float ic = 1.f / fmaxf(pcnt[g], 1.f);
    float p[32];
    #pragma unroll
    for (int o = 0; o < 32; ++o) p[o] = psum[g * 32 + o] * ic;
    float acc = r2b[0];
    #pragma unroll
    for (int j = 0; j < 16; ++j) {
        float s = r1b[j];
        #pragma unroll
        for (int o = 0; o < 32; ++o) s = fmaf(p[o], r1w[o * 16 + j], s);
        acc = fmaf(fmaxf(s, 0.f), r2w[j], acc);
    }
    out[g] = acc;
}

extern "C" void kernel_launch(void* const* d_in, const int* in_sizes, int n_in,
                              void* d_out, int out_size, void* d_ws, size_t ws_size,
                              hipStream_t stream) {
    const float* x     = (const float*)d_in[0];
    const float* ea    = (const float*)d_in[1];
    const float* e1w1  = (const float*)d_in[2];
    const float* e1b1  = (const float*)d_in[3];
    const float* e1w2  = (const float*)d_in[4];
    const float* e1b2  = (const float*)d_in[5];
    const float* root1 = (const float*)d_in[6];
    const float* bias1 = (const float*)d_in[7];
    const float* e2w1  = (const float*)d_in[8];
    const float* e2b1  = (const float*)d_in[9];
    const float* e2w2  = (const float*)d_in[10];
    const float* e2b2  = (const float*)d_in[11];
    const float* root2 = (const float*)d_in[12];
    const float* bias2 = (const float*)d_in[13];
    const float* bn1g  = (const float*)d_in[14];
    const float* bn1b  = (const float*)d_in[15];
    const float* bn1m  = (const float*)d_in[16];
    const float* bn1v  = (const float*)d_in[17];
    const float* bn2g  = (const float*)d_in[18];
    const float* bn2b  = (const float*)d_in[19];
    const float* bn2m  = (const float*)d_in[20];
    const float* bn2v  = (const float*)d_in[21];
    const float* r1w   = (const float*)d_in[22];
    const float* r1b   = (const float*)d_in[23];
    const float* r2w   = (const float*)d_in[24];
    const float* r2b   = (const float*)d_in[25];
    const int*   ei    = (const int*)d_in[26];
    const int*   batch = (const int*)d_in[27];

    char* ws = (char*)d_ws;
    unsigned* pB   = (unsigned*)(ws + 0);          //  2*16896 dwords = 135168 B
    unsigned* h1   = (unsigned*)(ws + 135168);     //  E*16 dwords   = 6.4 MB
    unsigned* h2   = (unsigned*)(ws + 6535168);    //  6.4 MB
    float*    agg1 = (float*)(ws + 12935168);      //  N*32 f32 = 1.28 MB
    float*    agg2 = (float*)(ws + 14215168);      //  1.28 MB
    float*    cnt  = (float*)(ws + 15495168);      //  40 KB
    float*    psum = (float*)(ws + 15535168);      //  8 KB
    float*    pcnt = (float*)(ws + 15543360);      //  256 B
    float*    x1   = (float*)(ws + 15543616);      //  1.28 MB
    float*    hfin = (float*)(ws + 16823616);      //  1.28 MB

    const int ZN = 652112;                         // agg1+agg2+cnt+psum+pcnt floats (contiguous)
    k_zero  <<<(ZN + 255) / 256, 256, 0, stream>>>(agg1, ZN);
    k_pack  <<<(2 * 16896 + 255) / 256, 256, 0, stream>>>(e1w2, e1b2, e2w2, e2b2, pB);
    k_counts<<<(E_ + 255) / 256, 256, 0, stream>>>(ei + E_, batch, cnt, pcnt);
    k_edge_h<<<(E_ + 255) / 256, 256, 0, stream>>>(ea, e1w1, e1b1, e2w1, e2b1, h1, h2);

    k_nnconv<<<160, 256, 66 * 1024, stream>>>(x, h1, pB, ei, agg1);
    k_node  <<<(N_ + 255) / 256, 256, 0, stream>>>(agg1, cnt, x, root1, bias1,
                                                   bn1g, bn1b, bn1m, bn1v, x1, batch, psum, 0);
    k_nnconv<<<160, 256, 66 * 1024, stream>>>(x1, h2, pB + 16896, ei, agg2);
    k_node  <<<(N_ + 255) / 256, 256, 0, stream>>>(agg2, cnt, x1, root2, bias2,
                                                   bn2g, bn2b, bn2m, bn2v, hfin, batch, psum, 1);
    k_readout<<<1, 64, 0, stream>>>(psum, pcnt, r1w, r1b, r2w, r2b, (float*)d_out);
}
